// MeanShift_22883585753208
// MI455X (gfx1250) — matrix operand order repaired and checked by reading, not compiled, not run on hardware
//
#include <hip/hip_runtime.h>

// ---------------------------------------------------------------------------
// MeanShift loss on MI455X (gfx1250, wave32, WMMA bf16)
//   B=4096, K=32768, DIM=512, TOPK=5
//   1) ms_norm   : L2-normalize rows -> bf16 (queryN, qbank; qbank[0:B]=target)
//   2) ms_main   : 16-row M-tile per block; 4 n-tiles x 2 matrices register-
//                  blocked WMMA bf16 stream over all K, B fragments double-
//                  buffered across k-steps; fused running top-5
//   3) ms_reduce : mean over per-row (2 - 0.4*sum(sim_q_top5))
// ---------------------------------------------------------------------------

typedef __attribute__((ext_vector_type(16))) __bf16        v16bf;
typedef __attribute__((ext_vector_type(8)))  float         v8f;
typedef __attribute__((ext_vector_type(4)))  unsigned int  u32x4;

union FragBF { u32x4 u[2]; v16bf v; };

#define MSB   4096
#define MSK   32768
#define MSDIM 512
#define ASTR  520      // LDS A row stride in ushorts (1040 B, 16B-multiple)
#define NT    4        // n-tiles per wave per iteration (register blocking)

__device__ __forceinline__ unsigned int f2bf(float x) {
    unsigned int u = __float_as_uint(x);
    u += 0x7FFFu + ((u >> 16) & 1u);            // round-to-nearest-even
    return u >> 16;
}
__device__ __forceinline__ unsigned int pk2bf(float a, float b) {
    return f2bf(a) | (f2bf(b) << 16);
}

// --------------------------- 1) normalize + convert -------------------------
__global__ __launch_bounds__(256) void ms_norm(
        const float* __restrict__ query_raw,
        const float* __restrict__ target_raw,
        const float* __restrict__ queue,
        unsigned short* __restrict__ queryN,
        unsigned short* __restrict__ qbank)
{
    const int w    = blockIdx.x * 8 + (threadIdx.x >> 5);   // one wave per row
    const int lane = threadIdx.x & 31;

    const float* src;
    unsigned short* dst;
    if (w < MSB) {
        src = query_raw + (size_t)w * MSDIM;
        dst = queryN    + (size_t)w * MSDIM;
    } else {
        const int r = w - MSB;        // r in [0, K)
        src = (w < 2 * MSB) ? (target_raw + (size_t)r * MSDIM)
                            : (queue      + (size_t)r * MSDIM);
        dst = qbank + (size_t)r * MSDIM;
    }

    float4 v[4];
    float ss = 0.f;
#pragma unroll
    for (int i = 0; i < 4; ++i) {
        v[i] = ((const float4*)src)[lane + i * 32];
        ss += v[i].x * v[i].x + v[i].y * v[i].y + v[i].z * v[i].z + v[i].w * v[i].w;
    }
#pragma unroll
    for (int off = 16; off > 0; off >>= 1) ss += __shfl_xor(ss, off, 32);
    const float sc = rsqrtf(ss);

#pragma unroll
    for (int i = 0; i < 4; ++i) {
        uint2 o;
        o.x = pk2bf(v[i].x * sc, v[i].y * sc);
        o.y = pk2bf(v[i].z * sc, v[i].w * sc);
        ((uint2*)dst)[lane + i * 32] = o;
    }
}

// ------------------- 2) fused dual-GEMM + running top-5 ---------------------
__global__ __launch_bounds__(256) void ms_main(
        const unsigned short* __restrict__ qbank,   // [K][512] bf16
        const unsigned short* __restrict__ queryN,  // [B][512] bf16
        float* __restrict__ rowsum)                 // [B]
{
    __shared__ unsigned short sAT[16 * ASTR];   // target A-tile (== qbank rows)
    __shared__ unsigned short sAQ[16 * ASTR];   // query  A-tile
    __shared__ float sSim[8][512];              // per-wave: [0:256)=simT, rest simQ
    __shared__ float sCand[16][160];            // 80 (key,payload) pairs per row

    const int tid  = threadIdx.x;
    const int wv   = tid >> 5;
    const int lane = tid & 31;
    const int half = lane >> 4;       // 0/1
    const int ln   = lane & 15;
    const int rowBase = blockIdx.x * 16;

    // ---- stage A tiles into LDS via async global->LDS (ASYNCcnt) ----
    for (int c = tid; c < 16 * 64; c += 256) {     // 16-byte chunks
        const int r = c >> 6, col = c & 63;
        const unsigned ldsT = (unsigned)(size_t)&sAT[r * ASTR + col * 8];
        const unsigned ldsQ = (unsigned)(size_t)&sAQ[r * ASTR + col * 8];
        const unsigned goff = (unsigned)(((rowBase + r) * MSDIM + col * 8) * 2);
        asm volatile("global_load_async_to_lds_b128 %0, %1, %2"
                     :: "v"(ldsT), "v"(goff),
                        "s"((unsigned long long)(const void*)qbank)
                     : "memory");
        asm volatile("global_load_async_to_lds_b128 %0, %1, %2"
                     :: "v"(ldsQ), "v"(goff),
                        "s"((unsigned long long)(const void*)queryN)
                     : "memory");
    }
    asm volatile("s_wait_asynccnt 0" ::: "memory");
    __syncthreads();

    // running top-5 (largest simT == smallest distT), payload = simQ
    float key[5], pay[5];
#pragma unroll
    for (int i = 0; i < 5; ++i) { key[i] = -1e30f; pay[i] = 0.f; }

    float* const st = &sSim[wv][0];

    for (int ntb = wv * NT; ntb < MSK / 16; ntb += 8 * NT) {
        // B-matrix 32x16 fragment source: lane ln holds bank row; lanes 0-15
        // cover K=0..15, lanes 16-31 K=16..31 (32 contiguous bytes / k-step)
        const unsigned short* brow[NT];
#pragma unroll
        for (int t = 0; t < NT; ++t)
            brow[t] = qbank + (size_t)((ntb + t) * 16 + ln) * MSDIM + half * 16;
        __builtin_prefetch(brow[0] + (size_t)8 * NT * 16 * MSDIM, 0, 0);
        __builtin_prefetch(brow[2] + (size_t)8 * NT * 16 * MSDIM, 0, 0);

        v8f accT[NT], accQ[NT];
#pragma unroll
        for (int t = 0; t < NT; ++t) { accT[t] = {}; accQ[t] = {}; }

        // double-buffered B fragments: prime kk=0
        FragBF bbuf[2][NT];
#pragma unroll
        for (int t = 0; t < NT; ++t) {
            const u32x4* bp = (const u32x4*)(brow[t]);
            bbuf[0][t].u[0] = bp[0];
            bbuf[0][t].u[1] = bp[1];
        }

#pragma unroll
        for (int kk = 0; kk < 16; ++kk) {
            const int cur = kk & 1, nxt = cur ^ 1;
            // issue next k-step's B loads first: one full k-step of slack
            if (kk < 15) {
#pragma unroll
                for (int t = 0; t < NT; ++t) {
                    const u32x4* bp = (const u32x4*)(brow[t] + (kk + 1) * 32);
                    bbuf[nxt][t].u[0] = bp[0];
                    bbuf[nxt][t].u[1] = bp[1];
                }
            }
            FragBF a, q;
            const int ab = ln * ASTR + kk * 32 + half * 16;  // ushort index
            a.u[0] = *(const u32x4*)&sAT[ab];
            a.u[1] = *(const u32x4*)&sAT[ab + 8];
            q.u[0] = *(const u32x4*)&sAQ[ab];
            q.u[1] = *(const u32x4*)&sAQ[ab + 8];
#pragma unroll
            for (int t = 0; t < NT; ++t) {
                accT[t] = __builtin_amdgcn_wmma_f32_16x16x32_bf16(
                              false, a.v, false, bbuf[cur][t].v, (short)0,
                              accT[t], false, false);
                accQ[t] = __builtin_amdgcn_wmma_f32_16x16x32_bf16(
                              false, q.v, false, bbuf[cur][t].v, (short)0,
                              accQ[t], false, false);
            }
        }

        // per-tile: bounce C tiles through per-wave LDS, then guarded top-5 scan
#pragma unroll
        for (int t = 0; t < NT; ++t) {
#pragma unroll
            for (int j = 0; j < 8; ++j) {
                st[(j + 8 * half) * 16 + ln]       = accT[t][j];
                st[256 + (j + 8 * half) * 16 + ln] = accQ[t][j];
            }
            asm volatile("s_wait_dscnt 0" ::: "memory");   // same-wave LDS RAW pin

#pragma unroll
            for (int i = 0; i < 8; ++i) {
                const int n = half * 8 + i;
                const float tv = st[ln * 16 + n];
                const float qv = st[256 + ln * 16 + n];
                if (tv > key[4]) {
                    float nk = tv, np = qv;
#pragma unroll
                    for (int s = 0; s < 5; ++s) {
                        const bool g = nk > key[s];
                        const float tk = g ? key[s] : nk;
                        const float tp = g ? pay[s] : np;
                        key[s] = g ? nk : key[s];
                        pay[s] = g ? np : pay[s];
                        nk = tk; np = tp;
                    }
                }
            }
            asm volatile("s_wait_dscnt 0" ::: "memory");   // before next tile's stores
        }
    }

    // ---- merge 8 waves x 2 halves x 5 = 80 candidates per row ----
    {
        const int base = wv * 10 + half * 5;
#pragma unroll
        for (int i = 0; i < 5; ++i) {
            sCand[ln][(base + i) * 2]     = key[i];
            sCand[ln][(base + i) * 2 + 1] = pay[i];
        }
    }
    __syncthreads();

    if (tid < 16) {
        float k5[5], p5[5];
#pragma unroll
        for (int i = 0; i < 5; ++i) { k5[i] = -1e30f; p5[i] = 0.f; }
        for (int c = 0; c < 80; ++c) {
            const float tv = sCand[tid][2 * c];
            const float qv = sCand[tid][2 * c + 1];
            if (tv > k5[4]) {
                float nk = tv, np = qv;
#pragma unroll
                for (int s = 0; s < 5; ++s) {
                    const bool g = nk > k5[s];
                    const float tk = g ? k5[s] : nk;
                    const float tp = g ? p5[s] : np;
                    k5[s] = g ? nk : k5[s];
                    p5[s] = g ? np : p5[s];
                    nk = tk; np = tp;
                }
            }
        }
        const float s5 = p5[0] + p5[1] + p5[2] + p5[3] + p5[4];
        // mean over top-5 of dist_q = 2 - 2*simQ  ->  2 - 0.4*sum(simQ)
        rowsum[rowBase + tid] = 2.0f - 0.4f * s5;
    }
}

// ----------------------------- 3) final mean --------------------------------
__global__ __launch_bounds__(256) void ms_reduce(
        const float* __restrict__ rowsum, float* __restrict__ out)
{
    __shared__ float sred[8];
    const int tid = threadIdx.x;
    float s = 0.f;
    for (int i = tid; i < MSB; i += 256) s += rowsum[i];
#pragma unroll
    for (int off = 16; off > 0; off >>= 1) s += __shfl_xor(s, off, 32);
    if ((tid & 31) == 0) sred[tid >> 5] = s;
    __syncthreads();
    if (tid == 0) {
        float v = 0.f;
#pragma unroll
        for (int i = 0; i < 8; ++i) v += sred[i];
        out[0] = v * (1.0f / (float)MSB);
    }
}

// ----------------------------------------------------------------------------
extern "C" void kernel_launch(void* const* d_in, const int* in_sizes, int n_in,
                              void* d_out, int out_size, void* d_ws, size_t ws_size,
                              hipStream_t stream)
{
    (void)in_sizes; (void)n_in; (void)out_size; (void)ws_size;
    const float* query_raw  = (const float*)d_in[0];
    const float* target_raw = (const float*)d_in[1];
    const float* queue      = (const float*)d_in[2];

    // ws: qbank bf16 [K][512] (32MB) | queryN bf16 [B][512] (4MB) | rowsum f32 [B]
    unsigned short* qbank  = (unsigned short*)d_ws;
    unsigned short* queryN = qbank + (size_t)MSK * MSDIM;
    float*          rowsum = (float*)(queryN + (size_t)MSB * MSDIM);
    float*          out    = (float*)d_out;

    ms_norm<<<(MSB + MSK) / 8, 256, 0, stream>>>(query_raw, target_raw, queue,
                                                 queryN, qbank);
    ms_main<<<MSB / 16, 256, 0, stream>>>(qbank, queryN, rowsum);
    ms_reduce<<<1, 256, 0, stream>>>(rowsum, out);
}
